// MyScaledDotProductWithBoxAttentionOne_65566970741302
// MI455X (gfx1250) — matrix operand order repaired and checked by reading, not compiled, not run on hardware
//
#include <hip/hip_runtime.h>

#define BATCH 16
#define NQ 512
#define NK 512
#define DM 512
#define HH 8
#define DH 64
#define MROWS (BATCH * NQ)   // 8192

typedef __attribute__((ext_vector_type(16))) __bf16 v16bf;
typedef __attribute__((ext_vector_type(8)))  float  v8f;

enum { OUT_HEADS = 0, OUT_KT = 1, OUT_F32 = 2 };

// ---------------------------------------------------------------- helpers

static __device__ __forceinline__ __bf16 to_bf16(float f) {
  union { float f; unsigned u; } a; a.f = f;
  unsigned r = a.u + 0x7FFFu + ((a.u >> 16) & 1u);   // round-to-nearest-even
  unsigned short h = (unsigned short)(r >> 16);
  __bf16 o; __builtin_memcpy(&o, &h, 2); return o;
}

// Async 16-byte global->LDS copy (CDNA5 GLOBAL_LOAD_ASYNC_TO_LDS_B128).
// LDS aperture maps flat addr[31:0] directly to the DS address (ISA 10.2),
// so the low 32 bits of a generic pointer to __shared__ are the LDS address.
static __device__ __forceinline__ void async_copy16(void* lds_ptr, const void* gptr) {
  unsigned lds = (unsigned)(unsigned long long)lds_ptr;
  unsigned long long ga = (unsigned long long)gptr;
  asm volatile("global_load_async_to_lds_b128 %0, %1, off"
               :: "v"(lds), "v"(ga) : "memory");
}
static __device__ __forceinline__ void async_wait0() {
  asm volatile("s_wait_asynccnt 0x0" ::: "memory");
}

// A 16x32 bf16 fragment from row-major tile As[m][k] (ld elements).
// Lane L: row = m0 + L%16; VGPRs 0-3 hold K = (L/16)*8 + 0..7 (contiguous),
// VGPRs 4-7 hold K = 16 + (L/16)*8 + 0..7  ->  2x ds_load_b128 per lane.
static __device__ __forceinline__ v16bf load_a_frag(const __bf16* As, int ld,
                                                    int m0, int kk, int lane) {
  int row  = m0 + (lane & 15);
  int half = lane >> 4;
  v16bf a;
#pragma unroll
  for (int e = 0; e < 8; ++e) a[e]     = As[row * ld + kk + half * 8 + e];
#pragma unroll
  for (int e = 0; e < 8; ++e) a[8 + e] = As[row * ld + kk + 16 + half * 8 + e];
  return a;
}

// B 32x16 bf16 fragment from an N-major tile Bt[n][k] (ld elements):
// lane L holds column n0 + L%16, K = kk + (L/16)*8 + 0..7 and kk + 16 + (L/16)*8 + 0..7
// -> also 2x ds_load_b128 per lane.
static __device__ __forceinline__ v16bf load_bt_frag(const __bf16* Bt, int ld,
                                                     int n0, int kk, int lane) {
  int col = n0 + (lane & 15);
  int ks  = kk + (lane >> 4) * 8;
  v16bf b;
#pragma unroll
  for (int e = 0; e < 16; ++e) b[e] = Bt[col * ld + ks + (e & 7) + (e >> 3) * 16];
  return b;
}

static __device__ __forceinline__ v8f zero8() {
  v8f z = {0.f, 0.f, 0.f, 0.f, 0.f, 0.f, 0.f, 0.f};
  return z;
}

// per-lane pv[r] holds partial for row (lane>>4)*8+r over this wave's columns;
// reduce across 16 lanes (shfl) then across 4 waves (LDS). red = float[64].
static __device__ __forceinline__ void row_reduce(float (&pv)[8], bool is_max,
                                                  float* red, int lane, int wid) {
#pragma unroll
  for (int r = 0; r < 8; ++r) {
    float v = pv[r];
#pragma unroll
    for (int m = 1; m < 16; m <<= 1) {
      float o = __shfl_xor(v, m, 32);
      v = is_max ? fmaxf(v, o) : (v + o);
    }
    pv[r] = v;
  }
  if ((lane & 15) == 0) {
#pragma unroll
    for (int r = 0; r < 8; ++r)
      red[((lane >> 4) * 8 + r) * 4 + wid] = pv[r];
  }
  __syncthreads();
#pragma unroll
  for (int r = 0; r < 8; ++r) {
    int row = (lane >> 4) * 8 + r;
    float v = red[row * 4 + 0];
#pragma unroll
    for (int q = 1; q < 4; ++q) {
      float o = red[row * 4 + q];
      v = is_max ? fmaxf(v, o) : (v + o);
    }
    pv[r] = v;
  }
  __syncthreads();
}

// ---------------------------------------------------------------- kernels

__global__ void f2bf_kernel(const float* __restrict__ x, __bf16* __restrict__ y, int n) {
  int i = blockIdx.x * 256 + threadIdx.x;
  if (i < n) y[i] = to_bf16(x[i]);
}

// 512x512 fp32 -> transposed bf16 (weights become N-major for B fragments)
__global__ void f2bf_t_kernel(const float* __restrict__ x, __bf16* __restrict__ y) {
  int i = blockIdx.x * 256 + threadIdx.x;   // 262144 total
  int r = i >> 9, c = i & 511;
  y[(size_t)c * DM + r] = to_bf16(x[(size_t)r * DM + c]);
}

// Generic GEMM: out = X[8192x512] @ W[512x512] + bias, W given TRANSPOSED (Wt[n][k]).
// Block tile 128x128, 8 waves (2x4), each wave 64x32 = 4x2 WMMA tiles.
// Tiles staged with GLOBAL_LOAD_ASYNC_TO_LDS_B128 (no VGPR round-trip).
__global__ __launch_bounds__(256)
void gemm_xw_kernel(const __bf16* __restrict__ X, const __bf16* __restrict__ Wt,
                    const float* __restrict__ bias, void* out, int outMode) {
  const int tid = threadIdx.x;
  const int lane = tid & 31;
  const int wid = tid >> 5;
  const int waveRow = wid >> 2;   // 0..1
  const int waveCol = wid & 3;    // 0..3
  const int bm = blockIdx.x * 128;
  const int bn = blockIdx.y * 128;

  __shared__ __bf16 As[128][40];   // [m][k], 32 k + pad
  __shared__ __bf16 Bs[128][40];   // [n][k], 32 k + pad (N-major!)

  v8f acc[4][2];
#pragma unroll
  for (int i = 0; i < 4; ++i)
#pragma unroll
    for (int j = 0; j < 2; ++j) acc[i][j] = zero8();

  for (int kk = 0; kk < DM; kk += 32) {
    __syncthreads();   // prior-iteration fragment reads are complete here
#pragma unroll
    for (int it = 0; it < 2; ++it) {               // A: 128x32, 16B chunks
      int idx = tid + it * 256;
      int row = idx >> 2, c = idx & 3;
      async_copy16(&As[row][c * 8], X + (size_t)(bm + row) * DM + kk + c * 8);
    }
#pragma unroll
    for (int it = 0; it < 2; ++it) {               // B: Wt rows n, 32 contiguous k
      int idx = tid + it * 256;
      int row = idx >> 2, c = idx & 3;
      async_copy16(&Bs[row][c * 8], Wt + (size_t)(bn + row) * DM + kk + c * 8);
    }
    async_wait0();
    __syncthreads();

    v16bf afr[4];
#pragma unroll
    for (int i = 0; i < 4; ++i)
      afr[i] = load_a_frag(&As[0][0], 40, waveRow * 64 + i * 16, 0, lane);
#pragma unroll
    for (int j = 0; j < 2; ++j) {
      v16bf bfr = load_bt_frag(&Bs[0][0], 40, waveCol * 32 + j * 16, 0, lane);
#pragma unroll
      for (int i = 0; i < 4; ++i)
        acc[i][j] = __builtin_amdgcn_wmma_f32_16x16x32_bf16(
            false, afr[i], false, bfr, (short)0, acc[i][j], false, false);
    }
  }

#pragma unroll
  for (int i = 0; i < 4; ++i) {
#pragma unroll
    for (int j = 0; j < 2; ++j) {
      int mbase = bm + waveRow * 64 + i * 16 + (lane >> 4) * 8;
      int col = bn + waveCol * 32 + j * 16 + (lane & 15);
      float bv = bias[col];
#pragma unroll
      for (int r = 0; r < 8; ++r) {
        int m = mbase + r;
        float val = acc[i][j][r] + bv;
        if (outMode == OUT_F32) {
          ((float*)out)[(size_t)m * DM + col] = val;
        } else {
          int bb = m >> 9, n = m & 511, h = col >> 6, d = col & 63;
          size_t off = (outMode == OUT_HEADS)
                           ? ((size_t)((bb * HH + h) * NQ + n) * DH + d)
                           : ((size_t)((bb * HH + h) * DH + d) * NK + n);
          ((__bf16*)out)[off] = to_bf16(val);
        }
      }
    }
  }
}

// Fused scores: S1=Q K1^T, S2=Q K2^T, scale*attw, mask, dual softmax, mix.
// One workgroup (4 waves) per (b*H+h, 16 q-rows). K stored [b,h,n,d]: the
// B operand is N-major (d = GEMM-K contiguous) -> b128 fragment loads.
// acc index j: global col = (j/4)*256 + wid*64 + (j%4)*16 + lane%16.
__global__ __launch_bounds__(128)
void attn_kernel(const __bf16* __restrict__ Qh,   // [B,H,NQ,DH]
                 const __bf16* __restrict__ K1,   // [B,H,NK,DH]
                 const __bf16* __restrict__ K2,
                 const float* __restrict__ attw,  // [B,H,NQ,NK]
                 const int* __restrict__ mask1,   // [B,NK]
                 const int* __restrict__ mask2,
                 __bf16* __restrict__ mix1,       // [B,H,NQ,NK]
                 __bf16* __restrict__ mix2) {
  const int tid = threadIdx.x;
  const int lane = tid & 31;
  const int wid = tid >> 5;         // 0..3
  const int bh = blockIdx.x;        // 0..127
  const int b = bh >> 3;
  const int q0 = blockIdx.y * 16;

  __shared__ __bf16 Qs[16][72];
  __shared__ __bf16 Ks[256][72];    // [n][d] chunk, N-major
  __shared__ float red[64];

  {  // load 16x64 Q block (128 chunks of 16B, one per thread)
    int row = tid >> 3, c = tid & 7;
    async_copy16(&Qs[row][c * 8],
                 Qh + ((size_t)bh * NQ + q0 + row) * DH + c * 8);
  }
  async_wait0();
  __syncthreads();
  v16bf afr[2];
  afr[0] = load_a_frag(&Qs[0][0], 72, 0, 0, lane);
  afr[1] = load_a_frag(&Qs[0][0], 72, 0, 32, lane);

  v8f acc[2][8];
#pragma unroll
  for (int t = 0; t < 2; ++t)
#pragma unroll
    for (int j = 0; j < 8; ++j) acc[t][j] = zero8();

#pragma unroll
  for (int t = 0; t < 2; ++t) {
    const __bf16* Kp = t ? K2 : K1;
#pragma unroll
    for (int nc = 0; nc < 2; ++nc) {
      __syncthreads();
#pragma unroll
      for (int it = 0; it < 16; ++it) {  // 256 rows x 64 d = 2048 x 16B
        int idx = tid + it * 128;
        int row = idx >> 3, c = idx & 7;
        async_copy16(&Ks[row][c * 8],
                     Kp + ((size_t)bh * NK + nc * 256 + row) * DH + c * 8);
      }
      async_wait0();
      __syncthreads();
#pragma unroll
      for (int jj = 0; jj < 4; ++jj) {
#pragma unroll
        for (int kh = 0; kh < 2; ++kh) {
          v16bf bfr = load_bt_frag(&Ks[0][0], 72, wid * 64 + jj * 16, kh * 32, lane);
          acc[t][nc * 4 + jj] = __builtin_amdgcn_wmma_f32_16x16x32_bf16(
              false, afr[kh], false, bfr, (short)0, acc[t][nc * 4 + jj], false, false);
        }
      }
    }
  }

  const float scale = 0.125f;  // 1/sqrt(64)
  const float NEG = -3.0e38f;
#pragma unroll
  for (int j = 0; j < 8; ++j) {
    int col = (j >> 2) * 256 + wid * 64 + (j & 3) * 16 + (lane & 15);
    int m1 = mask1[b * NK + col];
    int m2 = mask2[b * NK + col];
    const float* awp = attw + ((size_t)bh * NQ + q0) * NK + col;
#pragma unroll
    for (int r = 0; r < 8; ++r) {
      int mrow = (lane >> 4) * 8 + r;
      float aw = awp[(size_t)mrow * NK];
      float v1 = acc[0][j][r] * scale * aw;
      float v2 = acc[1][j][r] * scale * aw;
      acc[0][j][r] = m1 ? NEG : v1;
      acc[1][j][r] = m2 ? NEG : v2;
    }
  }

  float mx[2][8], sm[2][8];
#pragma unroll
  for (int t = 0; t < 2; ++t) {
    float pv[8];
#pragma unroll
    for (int r = 0; r < 8; ++r) {
      float v = acc[t][0][r];
#pragma unroll
      for (int j = 1; j < 8; ++j) v = fmaxf(v, acc[t][j][r]);
      pv[r] = v;
    }
    row_reduce(pv, true, red, lane, wid);
#pragma unroll
    for (int r = 0; r < 8; ++r) mx[t][r] = pv[r];
  }
#pragma unroll
  for (int t = 0; t < 2; ++t) {
#pragma unroll
    for (int j = 0; j < 8; ++j)
#pragma unroll
      for (int r = 0; r < 8; ++r)
        acc[t][j][r] = __expf(acc[t][j][r] - mx[t][r]);
    float pv[8];
#pragma unroll
    for (int r = 0; r < 8; ++r) {
      float v = 0.f;
#pragma unroll
      for (int j = 0; j < 8; ++j) v += acc[t][j][r];
      pv[r] = v;
    }
    row_reduce(pv, false, red, lane, wid);
#pragma unroll
    for (int r = 0; r < 8; ++r) sm[t][r] = pv[r];
  }

  float inv1[8], inv2[8];
#pragma unroll
  for (int r = 0; r < 8; ++r) { inv1[r] = 1.0f / sm[0][r]; inv2[r] = 1.0f / sm[1][r]; }

#pragma unroll
  for (int j = 0; j < 8; ++j) {
    int col = (j >> 2) * 256 + wid * 64 + (j & 3) * 16 + (lane & 15);
#pragma unroll
    for (int r = 0; r < 8; ++r) {
      int mrow = (lane >> 4) * 8 + r;
      float p1 = acc[0][j][r] * inv1[r];
      float p2 = acc[1][j][r] * inv2[r];
      size_t off = ((size_t)bh * NQ + q0 + mrow) * NK + col;
      mix1[off] = to_bf16(p1 + 0.1f * p2);
      mix2[off] = to_bf16(p2 + 0.1f * p1);
    }
  }
}

// out_heads[b, q, h*64+d] = mix[b,h,q,:] @ V[b,h,:,d]; V supplied TRANSPOSED
// (Vt[b,h,d,n], n = GEMM-K contiguous). One WG (4 waves) per (bh, 64 q-rows).
__global__ __launch_bounds__(128)
void av_kernel(const __bf16* __restrict__ mix,  // [B,H,NQ,NK]
               const __bf16* __restrict__ Vt,   // [B,H,DH,NK]
               __bf16* __restrict__ outh) {     // [B,NQ,H*DH]
  const int tid = threadIdx.x;
  const int lane = tid & 31;
  const int wid = tid >> 5;
  const int bh = blockIdx.x;
  const int b = bh >> 3, h = bh & 7;
  const int q0 = blockIdx.y * 64;

  __shared__ __bf16 As[64][40];   // [q][n-chunk]
  __shared__ __bf16 Bs[64][40];   // [d][n-chunk], N-major

  v8f acc[4];
#pragma unroll
  for (int j = 0; j < 4; ++j) acc[j] = zero8();

  for (int kk = 0; kk < NK; kk += 32) {
    __syncthreads();
#pragma unroll
    for (int it = 0; it < 2; ++it) {  // A 64x32 = 256 x 16B
      int idx = tid + it * 128;
      int row = idx >> 2, c = idx & 3;
      async_copy16(&As[row][c * 8],
                   mix + ((size_t)bh * NQ + q0 + row) * NK + kk + c * 8);
    }
#pragma unroll
    for (int it = 0; it < 2; ++it) {  // B: Vt rows d (64), 32 contiguous n
      int idx = tid + it * 128;
      int row = idx >> 2, c = idx & 3;
      async_copy16(&Bs[row][c * 8],
                   Vt + ((size_t)bh * DH + row) * NK + kk + c * 8);
    }
    async_wait0();
    __syncthreads();
    v16bf afr = load_a_frag(&As[0][0], 40, wid * 16, 0, lane);
#pragma unroll
    for (int j = 0; j < 4; ++j) {
      v16bf bfr = load_bt_frag(&Bs[0][0], 40, j * 16, 0, lane);
      acc[j] = __builtin_amdgcn_wmma_f32_16x16x32_bf16(
          false, afr, false, bfr, (short)0, acc[j], false, false);
    }
  }

#pragma unroll
  for (int j = 0; j < 4; ++j) {
#pragma unroll
    for (int r = 0; r < 8; ++r) {
      int q = q0 + wid * 16 + (lane >> 4) * 8 + r;
      int d = j * 16 + (lane & 15);
      outh[((size_t)b * NQ + q) * (HH * DH) + h * DH + d] = to_bf16(acc[j][r]);
    }
  }
}

// ---------------------------------------------------------------- host

extern "C" void kernel_launch(void* const* d_in, const int* in_sizes, int n_in,
                              void* d_out, int out_size, void* d_ws, size_t ws_size,
                              hipStream_t stream) {
  (void)in_sizes; (void)n_in; (void)out_size; (void)ws_size;

  const float* regions = (const float*)d_in[0];
  const float* grids   = (const float*)d_in[1];
  const float* inter   = (const float*)d_in[2];
  const int*   mask1   = (const int*)d_in[3];
  const int*   mask2   = (const int*)d_in[4];
  const float* attw    = (const float*)d_in[5];
  const float* Wq  = (const float*)d_in[6];  const float* bq  = (const float*)d_in[7];
  const float* Wk  = (const float*)d_in[8];  const float* bk  = (const float*)d_in[9];
  const float* Wv  = (const float*)d_in[10]; const float* bv  = (const float*)d_in[11];
  const float* Wo1 = (const float*)d_in[12]; const float* bo1 = (const float*)d_in[13];
  const float* Wo2 = (const float*)d_in[14]; const float* bo2 = (const float*)d_in[15];

  char* w = (char*)d_ws;
  auto alloc = [&](size_t bytes) {
    char* p = w; w += (bytes + 255) & ~(size_t)255; return p;
  };
  const size_t XE = (size_t)MROWS * DM;           // 8192*512
  const size_t WE = (size_t)DM * DM;              // 512*512
  const size_t SE = (size_t)BATCH * HH * NQ * NK; // 33.5M

  __bf16* Xr   = (__bf16*)alloc(XE * 2);
  __bf16* Xg   = (__bf16*)alloc(XE * 2);
  __bf16* Xi   = (__bf16*)alloc(XE * 2);
  __bf16* Wqt  = (__bf16*)alloc(WE * 2);   // transposed weights (N-major)
  __bf16* Wkt  = (__bf16*)alloc(WE * 2);
  __bf16* Wvt  = (__bf16*)alloc(WE * 2);
  __bf16* Wo1t = (__bf16*)alloc(WE * 2);
  __bf16* Wo2t = (__bf16*)alloc(WE * 2);
  __bf16* Qh   = (__bf16*)alloc(XE * 2);   // [B,H,NQ,DH]
  __bf16* K1   = (__bf16*)alloc(XE * 2);   // [B,H,NK,DH]
  __bf16* K2   = (__bf16*)alloc(XE * 2);
  __bf16* V1t  = (__bf16*)alloc(XE * 2);   // [B,H,DH,NK]
  __bf16* V2t  = (__bf16*)alloc(XE * 2);
  __bf16* mx1  = (__bf16*)alloc(SE * 2);
  __bf16* mx2  = (__bf16*)alloc(SE * 2);
  __bf16* oh1  = (__bf16*)alloc(XE * 2);
  __bf16* oh2  = (__bf16*)alloc(XE * 2);

  // stage 0: fp32 -> bf16 (weights transposed)
  {
    int nX = (int)XE;
    dim3 gX((nX + 255) / 256), gW((int)WE / 256), t(256);
    f2bf_kernel<<<gX, t, 0, stream>>>(regions, Xr, nX);
    f2bf_kernel<<<gX, t, 0, stream>>>(grids,   Xg, nX);
    f2bf_kernel<<<gX, t, 0, stream>>>(inter,   Xi, nX);
    f2bf_t_kernel<<<gW, t, 0, stream>>>(Wq,  Wqt);
    f2bf_t_kernel<<<gW, t, 0, stream>>>(Wk,  Wkt);
    f2bf_t_kernel<<<gW, t, 0, stream>>>(Wv,  Wvt);
    f2bf_t_kernel<<<gW, t, 0, stream>>>(Wo1, Wo1t);
    f2bf_t_kernel<<<gW, t, 0, stream>>>(Wo2, Wo2t);
  }

  // stage 1: projections (WMMA GEMMs). K natural layout, V transposed.
  {
    dim3 g(MROWS / 128, DM / 128), t(256);
    gemm_xw_kernel<<<g, t, 0, stream>>>(Xi, Wqt, bq, Qh,  OUT_HEADS);
    gemm_xw_kernel<<<g, t, 0, stream>>>(Xr, Wkt, bk, K1,  OUT_HEADS);
    gemm_xw_kernel<<<g, t, 0, stream>>>(Xg, Wkt, bk, K2,  OUT_HEADS);
    gemm_xw_kernel<<<g, t, 0, stream>>>(Xr, Wvt, bv, V1t, OUT_KT);
    gemm_xw_kernel<<<g, t, 0, stream>>>(Xg, Wvt, bv, V2t, OUT_KT);
  }

  // stage 2: fused dual scores + softmax + mix
  {
    dim3 g(BATCH * HH, NQ / 16), t(128);
    attn_kernel<<<g, t, 0, stream>>>(Qh, K1, K2, attw, mask1, mask2, mx1, mx2);
  }

  // stage 3: (att + 0.1 att') @ V
  {
    dim3 g(BATCH * HH, NQ / 64), t(128);
    av_kernel<<<g, t, 0, stream>>>(mx1, V1t, oh1);
    av_kernel<<<g, t, 0, stream>>>(mx2, V2t, oh2);
  }

  // stage 4: output projections, fp32 into d_out (out1 then out2)
  {
    dim3 g(MROWS / 128, DM / 128), t(256);
    gemm_xw_kernel<<<g, t, 0, stream>>>(oh1, Wo1t, bo1, d_out, OUT_F32);
    gemm_xw_kernel<<<g, t, 0, stream>>>(oh2, Wo2t, bo2,
                                        (void*)((float*)d_out + XE), OUT_F32);
  }
}
